// Mamba_28089086116520
// MI455X (gfx1250) — matrix operand order repaired and checked by reading, not compiled, not run on hardware
//
#include <hip/hip_runtime.h>
#include <hip/hip_bf16.h>

// ---------------------------------------------------------------------------
// Mamba-like selective scan, MI455X (gfx1250, wave32, WMMA bf16 16x16x32).
//   B=8, L=4096, D=512, H(=MAXD=HID)=128, chunks C=64 x T=64.
// Pipeline:
//   prep:   W_cc = W_conv@W_comp, W_ck = W_kernel@W_comp, W_f = W_decomp@W_out
//           written bf16, pre-swizzled into WMMA B-operand layout.
//   G1:     xc = silu(x @ W_ccT), k = x @ W_ckT          (WMMA)
//   S1/S2:  chunked parallel scan reduce + prefix        (VALU, coalesced)
//   S3G2:   scan replay -> state tile in LDS (A layout) -> y = state @ W_fT (WMMA)
// Cache policy: x loads / y stores non-temporal (streamed once);
//              xc/k intermediates regular-temporal (written once, read twice,
//              fits in 192MB L2).
// ---------------------------------------------------------------------------

typedef unsigned short u16;
typedef __attribute__((ext_vector_type(16))) __bf16 v16bf;
typedef __attribute__((ext_vector_type(8)))  float  v8f;
typedef __attribute__((ext_vector_type(4)))  float  v4f;

#define NB    8
#define LL    4096
#define DD    512
#define HH    128
#define NCH   64      // chunks per sequence
#define CT    64      // timesteps per chunk
#define BLROWS 32768  // B*L

__device__ __forceinline__ u16 f2bf(float f) {
    unsigned u = __float_as_uint(f);
    unsigned r = 0x7FFFu + ((u >> 16) & 1u);   // round-to-nearest-even
    return (u16)((u + r) >> 16);
}

__device__ __forceinline__ float silu_fast(float v) {
    // v * rcp(1+exp(-v)) : one v_exp_f32 + one v_rcp_f32, no IEEE div chain
    return v * __builtin_amdgcn_rcpf(1.0f + __expf(-v));
}

// ---------------------------------------------------------------------------
// prep: out[n,k] = sum_j A1[n,j] * B1[j,k]  (inner dim always 128),
// stored bf16 in WMMA B-operand swizzle:
//   kc=k>>5, kk=k&31 ; lane=(n&15)|(((kk>>3)&1)<<4) ; e=((kk>>4)<<3)|(kk&7) ; nt=n>>4
//   off = ((kc*NT + nt)*32 + lane)*16 + e
// ---------------------------------------------------------------------------
__global__ void prep_gemm_swz(const float* __restrict__ A1,
                              const float* __restrict__ B1,
                              u16* __restrict__ out, int Ndim, int Kdim) {
    int idx = blockIdx.x * blockDim.x + threadIdx.x;
    if (idx >= Ndim * Kdim) return;
    int n = idx / Kdim, k = idx - n * Kdim;
    float acc = 0.f;
#pragma unroll 8
    for (int j = 0; j < 128; ++j) acc += A1[n * 128 + j] * B1[j * Kdim + k];
    int NT = Ndim >> 4;
    int kc = k >> 5, kk = k & 31;
    int lane = (n & 15) | (((kk >> 3) & 1) << 4);
    int e = ((kk >> 4) << 3) | (kk & 7), nt = n >> 4;
    out[(((kc * NT + nt) * 32 + lane) << 4) + e] = f2bf(acc);
}

// ---------------------------------------------------------------------------
// G1: per block, 16 rows of x (16x512) -> xc(16x128, silu) and k(16x128).
// A staged to LDS in the ISA 16-bit A layout:
//   half=(kk>>3)&1 ; lane=m|(half<<4) ; e=((kk>>4)<<3)|(kk&7)
// 4 waves, each owns 4 N-tiles (waves 0-1 -> W_cc, waves 2-3 -> W_ck).
// Per K-chunk: issue all 4 B-tile loads as one clause, then 4 independent
// WMMA accumulation chains.
// ---------------------------------------------------------------------------
__global__ __launch_bounds__(128) void g1_kernel(const float* __restrict__ x,
        const u16* __restrict__ Wcc, const u16* __restrict__ Wck,
        float* __restrict__ xc, float* __restrict__ kout) {
    __shared__ __attribute__((aligned(32))) u16 lds_a[16 * 32 * 16];  // 16 KB
    const int tid = threadIdx.x;
    const long long bl0 = (long long)blockIdx.x * 16;

#pragma unroll 4
    for (int i = 0; i < 64; ++i) {            // 16 rows x 512 cols, coalesced
        int flat = tid + i * 128;
        int m = flat >> 9, d = flat & 511;
        float v = __builtin_nontemporal_load(&x[(bl0 + m) * DD + d]);
        int kc = d >> 5, kk = d & 31;
        int half = (kk >> 3) & 1;
        int lane = m | (half << 4);
        int e = ((kk >> 4) << 3) | (kk & 7);
        lds_a[((kc * 32 + lane) << 4) + e] = f2bf(v);
    }
    __syncthreads();

    const int wave = tid >> 5, lane = tid & 31;
    const u16* bm = (wave < 2) ? Wcc : Wck;    // waves 0-1: W_cc, 2-3: W_ck
    const int ntbase = (wave & 1) * 4;
    v8f acc[4] = {};
#pragma unroll 1
    for (int kc = 0; kc < 16; ++kc) {
        v16bf bv[4];
#pragma unroll
        for (int t = 0; t < 4; ++t)            // one load clause, overlap WMMAs
            bv[t] = *(const v16bf*)&bm[(((kc * 8 + (ntbase + t)) * 32 + lane) << 4)];
        v16bf a = *(const v16bf*)&lds_a[(kc * 32 + lane) << 4];
#pragma unroll
        for (int t = 0; t < 4; ++t)            // 4 independent acc chains
            acc[t] = __builtin_amdgcn_wmma_f32_16x16x32_bf16(
                false, a, false, bv[t], (short)0, acc[t], false, false);
    }
    const int n = lane & 15, half = lane >> 4;
    float* dst = (wave < 2) ? xc : kout;
    const bool is_cc = wave < 2;
#pragma unroll
    for (int t = 0; t < 4; ++t) {
        int nt = ntbase + t;
#pragma unroll
        for (int r = 0; r < 8; ++r) {
            int m = r + half * 8;               // C/D layout: lanes>=16 hold M=8..15
            float v = acc[t][r];
            if (is_cc) v = silu_fast(v);
            dst[(bl0 + m) * HH + nt * 16 + n] = v;
        }
    }
}

// ---------------------------------------------------------------------------
// S1: per (b,chunk,h) local reduce:  a = sum xc*p, P = prod k   (p starts 1)
// ---------------------------------------------------------------------------
__global__ __launch_bounds__(128) void s1_kernel(const float* __restrict__ xc,
        const float* __restrict__ kin,
        float* __restrict__ chunkP, float* __restrict__ chunkA) {
    const int bc = blockIdx.x;                  // b*64+c ; base row = bc*64
    const long long base = (long long)bc * CT;
    const int h = threadIdx.x;
    float p = 1.f, a = 0.f;
#pragma unroll 4
    for (int t = 0; t < CT; ++t) {
        long long idx = (base + t) * HH + h;
        a += xc[idx] * p;
        p *= kin[idx];
    }
    chunkP[(long long)bc * HH + h] = p;
    chunkA[(long long)bc * HH + h] = a;
}

// ---------------------------------------------------------------------------
// S2: exclusive prefix over the 64 chunks of each batch (composition of
//     affine maps: a0' = a0 + p0*A ; p0' = p0*P).
// ---------------------------------------------------------------------------
__global__ __launch_bounds__(128) void s2_kernel(const float* __restrict__ chunkP,
        const float* __restrict__ chunkA,
        float* __restrict__ prefP, float* __restrict__ prefA) {
    const int b = blockIdx.x, h = threadIdx.x;
    float p0 = 1.f, a0 = 0.f;
    for (int c = 0; c < NCH; ++c) {
        long long i = ((long long)(b * NCH + c)) * HH + h;
        prefP[i] = p0;
        prefA[i] = a0;
        a0 += p0 * chunkA[i];
        p0 *= chunkP[i];
    }
}

// ---------------------------------------------------------------------------
// S3+G2 fused: replay scan for one 64-row chunk, write state bf16 straight
// into LDS in the WMMA A layout (4 row-tiles x 4 K-chunks), then
// y[64,512] = state[64,128] @ W_fT. 4 waves x (16 x 2 N-tiles) x 4 K WMMAs,
// two independent accumulation chains per iteration for ILP.
// ---------------------------------------------------------------------------
__global__ __launch_bounds__(128) void s3g2_kernel(const float* __restrict__ xc,
        const float* __restrict__ kin, const float* __restrict__ prefP,
        const float* __restrict__ prefA, const u16* __restrict__ Wf,
        float* __restrict__ y) {
    __shared__ __attribute__((aligned(32))) u16 lds_s[4 * 4 * 32 * 16];  // 16 KB
    const int bc = blockIdx.x;
    const long long base = (long long)bc * CT;  // b*4096 + c*64 == bc*64
    const int h = threadIdx.x;

    float p = prefP[(long long)bc * HH + h];
    float a = prefA[(long long)bc * HH + h];
    const int kc = h >> 5, kk = h & 31;
    const int laneS = ((kk >> 3) & 1) << 4;
    const int e = ((kk >> 4) << 3) | (kk & 7);
#pragma unroll 4
    for (int t = 0; t < CT; ++t) {
        long long idx = (base + t) * HH + h;
        a += xc[idx] * p;                        // state value at t
        int rt = t >> 4, m = t & 15;
        lds_s[(((rt * 4 + kc) * 32 + (m | laneS)) << 4) + e] = f2bf(a);
        p *= kin[idx];
    }
    __syncthreads();

    const int wave = h >> 5, lane = h & 31;
    const int rt = wave;                         // one 16-row tile per wave
    v16bf av[4];
#pragma unroll
    for (int c4 = 0; c4 < 4; ++c4)
        av[c4] = *(const v16bf*)&lds_s[((rt * 4 + c4) * 32 + lane) << 4];

    const int n = lane & 15, halfo = lane >> 4;
#pragma unroll 1
    for (int nt = 0; nt < 32; nt += 2) {
        v16bf b0[4], b1[4];
#pragma unroll
        for (int c4 = 0; c4 < 4; ++c4) {         // batch all 8 B-tile loads
            b0[c4] = *(const v16bf*)&Wf[(((c4 * 32 + nt) * 32 + lane) << 4)];
            b1[c4] = *(const v16bf*)&Wf[(((c4 * 32 + nt + 1) * 32 + lane) << 4)];
        }
        v8f acc0 = {}, acc1 = {};
#pragma unroll
        for (int c4 = 0; c4 < 4; ++c4) {         // two independent chains
            acc0 = __builtin_amdgcn_wmma_f32_16x16x32_bf16(
                false, av[c4], false, b0[c4], (short)0, acc0, false, false);
            acc1 = __builtin_amdgcn_wmma_f32_16x16x32_bf16(
                false, av[c4], false, b1[c4], (short)0, acc1, false, false);
        }
#pragma unroll
        for (int r = 0; r < 8; ++r) {
            int m = r + halfo * 8;
            long long row = base + rt * 16 + m;
            __builtin_nontemporal_store(acc0[r], &y[row * DD + nt * 16 + n]);
            __builtin_nontemporal_store(acc1[r], &y[row * DD + (nt + 1) * 16 + n]);
        }
    }
}

// ---------------------------------------------------------------------------
extern "C" void kernel_launch(void* const* d_in, const int* in_sizes, int n_in,
                              void* d_out, int out_size, void* d_ws, size_t ws_size,
                              hipStream_t stream) {
    (void)in_sizes; (void)n_in; (void)out_size; (void)ws_size;
    const float* x        = (const float*)d_in[0];  // [8,4096,512]
    const float* W_comp   = (const float*)d_in[1];  // [128,512]
    const float* W_conv   = (const float*)d_in[2];  // [128,128]
    /* d_in[3] = W_state: state0 = zeros @ W_stateT == 0, unused */
    const float* W_kernel = (const float*)d_in[4];  // [128,128]
    const float* W_out    = (const float*)d_in[5];  // [128,128]
    const float* W_decomp = (const float*)d_in[6];  // [512,128]
    float* y = (float*)d_out;

    char* ws = (char*)d_ws;
    size_t o = 0;
    u16* Wcc = (u16*)(ws + o); o += (size_t)HH * DD * 2;        // 128 KB
    u16* Wck = (u16*)(ws + o); o += (size_t)HH * DD * 2;        // 128 KB
    u16* Wf  = (u16*)(ws + o); o += (size_t)DD * HH * 2;        // 128 KB
    float* xc     = (float*)(ws + o); o += (size_t)BLROWS * HH * 4;  // 16 MB
    float* kbuf   = (float*)(ws + o); o += (size_t)BLROWS * HH * 4;  // 16 MB
    float* chunkP = (float*)(ws + o); o += (size_t)NB * NCH * HH * 4;
    float* chunkA = (float*)(ws + o); o += (size_t)NB * NCH * HH * 4;
    float* prefP  = (float*)(ws + o); o += (size_t)NB * NCH * HH * 4;
    float* prefA  = (float*)(ws + o); o += (size_t)NB * NCH * HH * 4;

    // 1) fold + swizzle weights (N*K = 65536 elements each)
    prep_gemm_swz<<<256, 256, 0, stream>>>(W_conv,   W_comp, Wcc, HH, DD);
    prep_gemm_swz<<<256, 256, 0, stream>>>(W_kernel, W_comp, Wck, HH, DD);
    prep_gemm_swz<<<256, 256, 0, stream>>>(W_decomp, W_out,  Wf,  DD, HH);

    // 2) input GEMMs: xc (silu) and k
    g1_kernel<<<BLROWS / 16, 128, 0, stream>>>(x, Wcc, Wck, xc, kbuf);

    // 3) chunked scan: local reduce, cross-chunk prefix
    s1_kernel<<<NB * NCH, 128, 0, stream>>>(xc, kbuf, chunkP, chunkA);
    s2_kernel<<<NB, 128, 0, stream>>>(chunkP, chunkA, prefP, prefA);

    // 4) scan replay fused with output GEMM
    s3g2_kernel<<<NB * NCH, 128, 0, stream>>>(xc, kbuf, prefP, prefA, Wf, y);
}
